// FP8StaticLinearQuantizer_36799279792445
// MI455X (gfx1250) — 8-bit matrix operand order repaired and checked by reading, compiled, not run on hardware
//
#include <hip/hip_runtime.h>
#include <stdint.h>

typedef __attribute__((ext_vector_type(16))) int   v16i;
typedef __attribute__((ext_vector_type(4)))  int   v4i;
typedef __attribute__((ext_vector_type(8)))  float v8f;

#define FP8_MAX_F 448.0f
#define K_DIM 1024
#define N_DIM 1024

__global__ void ws_init_kernel(unsigned int* amax_bits) {
    if (threadIdx.x == 0 && blockIdx.x == 0) amax_bits[0] = 0u;
}

__global__ __launch_bounds__(256) void amax_kernel(const float* __restrict__ x,
                                                   unsigned int* __restrict__ amax_bits,
                                                   int n4) {
    int idx    = blockIdx.x * blockDim.x + threadIdx.x;
    int stride = gridDim.x * blockDim.x;
    const float4* x4 = (const float4*)x;
    float m = 0.0f;
    for (int i = idx; i < n4; i += stride) {
        float4 v = x4[i];
        m = fmaxf(m, fmaxf(fmaxf(fabsf(v.x), fabsf(v.y)),
                           fmaxf(fabsf(v.z), fabsf(v.w))));
    }
    #pragma unroll
    for (int off = 16; off >= 1; off >>= 1)
        m = fmaxf(m, __shfl_xor(m, off, 32));
    __shared__ float smax[8];
    int lane = threadIdx.x & 31, w = threadIdx.x >> 5;
    if (lane == 0) smax[w] = m;
    __syncthreads();
    if (w == 0) {
        m = (lane < 8) ? smax[lane] : 0.0f;
        #pragma unroll
        for (int off = 4; off >= 1; off >>= 1)
            m = fmaxf(m, __shfl_xor(m, off, 32));
        if (lane == 0) atomicMax(amax_bits, __float_as_uint(m));
    }
}

__device__ inline unsigned int sw_f32_to_e4m3(float f) {
    unsigned int u    = __float_as_uint(f);
    unsigned int sign = (u >> 24) & 0x80u;
    unsigned int a    = u & 0x7FFFFFFFu;
    if (a < 0x3C800000u) {
        float af = __uint_as_float(a);
        unsigned int m = (unsigned int)__builtin_rintf(af * 512.0f);
        return sign | m;
    }
    unsigned int exp   = a >> 23;
    unsigned int mant  = a & 0x7FFFFFu;
    unsigned int mant3 = (mant + 0x7FFFFu + ((mant >> 20) & 1u)) >> 20;
    if (mant3 == 8u) { mant3 = 0u; exp += 1u; }
    unsigned int e4 = exp - 127u + 7u;
    return sign | (e4 << 3) | mant3;
}

__device__ inline int pack4_fp8(float a, float b, float c, float d) {
#if __has_builtin(__builtin_amdgcn_cvt_pk_fp8_f32)
    int r = 0;
    r = __builtin_amdgcn_cvt_pk_fp8_f32(a, b, r, false);
    r = __builtin_amdgcn_cvt_pk_fp8_f32(c, d, r, true);
    return r;
#else
    return (int)(sw_f32_to_e4m3(a) | (sw_f32_to_e4m3(b) << 8) |
                 (sw_f32_to_e4m3(c) << 16) | (sw_f32_to_e4m3(d) << 24));
#endif
}

__global__ __launch_bounds__(256) void quant_kernel(const float* __restrict__ x,
                                                    unsigned char* __restrict__ q,
                                                    const unsigned int* __restrict__ amax_bits,
                                                    int n16) {
    float amax = __uint_as_float(*amax_bits);
    float s = FP8_MAX_F / fmaxf(amax, 1e-12f);
    int idx = blockIdx.x * blockDim.x + threadIdx.x;
    if (idx >= n16) return;
    const float4* x4 = (const float4*)x;
    int4 packed;
    int* pw = (int*)&packed;
    #pragma unroll
    for (int j = 0; j < 4; ++j) {
        float4 v = x4[idx * 4 + j];
        float a = fminf(fmaxf(v.x * s, -FP8_MAX_F), FP8_MAX_F);
        float b = fminf(fmaxf(v.y * s, -FP8_MAX_F), FP8_MAX_F);
        float c = fminf(fmaxf(v.z * s, -FP8_MAX_F), FP8_MAX_F);
        float d = fminf(fmaxf(v.w * s, -FP8_MAX_F), FP8_MAX_F);
        pw[j] = pack4_fp8(a, b, c, d);
    }
    ((int4*)q)[idx] = packed;
}

typedef __attribute__((address_space(1))) v4i* gv4i_ptr;
typedef __attribute__((address_space(3))) v4i* lv4i_ptr;

__device__ inline void async_load_b128(const unsigned char* g, unsigned char* l) {
#if __has_builtin(__builtin_amdgcn_global_load_async_to_lds_b128)
    __builtin_amdgcn_global_load_async_to_lds_b128(
        (gv4i_ptr)(v4i*)g, (lv4i_ptr)(v4i*)l, 0, 0);
#else
    lv4i_ptr l3 = (lv4i_ptr)(v4i*)l;
    gv4i_ptr g1 = (gv4i_ptr)(v4i*)g;
    asm volatile("global_load_async_to_lds_b128 %0, %1, off"
                 :: "v"(l3), "v"(g1) : "memory");
#endif
}

__device__ inline void wait_async0() {
#if __has_builtin(__builtin_amdgcn_s_wait_asynccnt)
    __builtin_amdgcn_s_wait_asynccnt(0);
#else
    asm volatile("s_wait_asynccnt 0x0" ::: "memory");
#endif
}

__global__ __launch_bounds__(256) void fp8_gemm_kernel(
        const unsigned char* __restrict__ A,
        const unsigned char* __restrict__ W,
        const unsigned int* __restrict__ amax_bits,
        const float* __restrict__ wscale,
        float* __restrict__ out) {
    __shared__ __align__(16) unsigned char lds_w[64 * K_DIM];

    const int tid  = threadIdx.x;
    const int lane = tid & 31;
    const int wave = tid >> 5;
    const int hi   = lane >> 4;
    const int lo   = lane & 15;

    const int m0 = blockIdx.x * 128 + wave * 16;
    const int n0 = blockIdx.y * 64;

    #pragma unroll
    for (int i = 0; i < 16; ++i) {
        const int q   = i * 256 + tid;
        const int ql  = q & 31;
        const int jb  = (q >> 5) & 3;
        const int nt  = (q >> 7) & 3;
        const int s   = q >> 9;
        const int row = nt * 16 + (ql & 15);
        const int col = s * 128 + (ql >> 4) * 16 + jb * 32;
        async_load_b128(W + (size_t)(n0 + row) * K_DIM + col, &lds_w[q << 4]);
    }
    wait_async0();
    __syncthreads();

    const float amax  = __uint_as_float(*amax_bits);
    const float scale = fmaxf(amax, 1e-12f) * (1.0f / FP8_MAX_F) * wscale[0];

    const unsigned char* Arow = A + (size_t)(m0 + lo) * K_DIM + hi * 16;
    const unsigned char* Brow = &lds_w[lane * 16];

    v8f acc[4] = {v8f{}, v8f{}, v8f{}, v8f{}};

    for (int s = 0; s < 8; ++s) {
        const int kk = s * 128;
        v16i a;
        #pragma unroll
        for (int j = 0; j < 8; ++j) {
            const int koff = (j >> 1) * 32 + (j & 1) * 8;
            const uint2 t = *(const uint2*)(Arow + kk + koff);
            a[2 * j]     = (int)t.x;
            a[2 * j + 1] = (int)t.y;
        }
        #pragma unroll
        for (int nt = 0; nt < 4; ++nt) {
            v16i b;
            #pragma unroll
            for (int jb = 0; jb < 4; ++jb) {
                const int4 t = *(const int4*)(Brow + ((s * 4 + nt) * 4 + jb) * 512);
                b[4 * jb]     = t.x;
                b[4 * jb + 1] = t.y;
                b[4 * jb + 2] = t.z;
                b[4 * jb + 3] = t.w;
            }
            acc[nt] = __builtin_amdgcn_wmma_f32_16x16x128_fp8_fp8(
                a, b, (short)0, acc[nt], false, false);
        }
    }

    #pragma unroll
    for (int nt = 0; nt < 4; ++nt) {
        const int n = n0 + nt * 16 + lo;
        #pragma unroll
        for (int r = 0; r < 8; ++r) {
            const int m = m0 + hi * 8 + r;
            __builtin_nontemporal_store(acc[nt][r] * scale,
                                        &out[(size_t)m * N_DIM + n]);
        }
    }
}

extern "C" void kernel_launch(void* const* d_in, const int* in_sizes, int n_in,
                              void* d_out, int out_size, void* d_ws, size_t ws_size,
                              hipStream_t stream) {
    const float*         x      = (const float*)d_in[0];
    const unsigned char* qw     = (const unsigned char*)d_in[1];
    const float*         wscale = (const float*)d_in[2];
    float*               out    = (float*)d_out;

    const int n_elem = in_sizes[0];
    const int M      = n_elem / K_DIM;

    unsigned int*  amax_bits = (unsigned int*)d_ws;
    unsigned char* qx        = (unsigned char*)d_ws + 256;

    ws_init_kernel<<<1, 64, 0, stream>>>(amax_bits);

    const int n4 = n_elem / 4;
    amax_kernel<<<2048, 256, 0, stream>>>(x, amax_bits, n4);

    const int n16 = n_elem / 16;
    quant_kernel<<<(n16 + 255) / 256, 256, 0, stream>>>(x, qx, amax_bits, n16);

    dim3 grid(M / 128, N_DIM / 64);
    fp8_gemm_kernel<<<grid, 256, 0, stream>>>(qx, qw, amax_bits, wscale, out);
}
